// Net_62517543961209
// MI455X (gfx1250) — matrix operand order repaired and checked by reading, not compiled, not run on hardware
//
#include <hip/hip_runtime.h>

// Deformable group correlation for MI455X (gfx1250, wave32, WMMA).
// N=2, C=128, H=64, W=256, GROUPS=8, S=9, cg=16.
//
// Memory/latency-bound op (~45MB unique traffic -> ~2us floor at 23.3 TB/s).
// Gather phase: per-pixel bilinear weights/indices hoisted out of the
// 128-channel loop; 4 always-in-bounds corner loads with edge weights
// pre-zeroed. Correlation phase: per 16-pixel tile the 8 group-wise
// 16-channel dots are the diagonal of a 16x16x32 f16 WMMA (K zero-padded).
// Zero padding lives in LDS so fragment loads are single 32B ds_loads with
// fully uniform control flow (no EXEC masking around WMMA).

typedef __attribute__((ext_vector_type(16))) _Float16 v16h;
typedef __attribute__((ext_vector_type(8)))  _Float16 v8h;
typedef __attribute__((ext_vector_type(8)))  float    v8f;

constexpr int NN = 2, CC = 128, HH = 64, WW = 256;
constexpr int GROUPS = 8, S = 9, CG = 16;
constexpr int TP   = 16;    // pixels per tile (WMMA M/N)
constexpr int AROW = 272;   // halves per pixel row of A-staging (544B = 17*32B)
constexpr int BROW = 144;   // halves per pixel row of B-staging (288B = 9*32B)
constexpr int DSTR = 18;    // dmat row stride (floats), bank-conflict free spill

__global__ __launch_bounds__(32)
void deform_corr_wmma(const float* __restrict__ left,
                      const float* __restrict__ right,
                      const float* __restrict__ eo,
                      float* __restrict__ out)
{
    // A staging: per (pixel, group) 32 halves laid out exactly as the
    // 16x32 f16 A fragment: [ch0..15 | zeros]
    __shared__ __align__(32) _Float16 aL[TP][AROW];
    // B staging: per pixel 128 contiguous halves (group-major channels)
    __shared__ __align__(32) _Float16 bL[TP][BROW];
    // 32B zero block: B fragment source for lanes 16..31 (K=16..31 pad)
    __shared__ __align__(32) _Float16 zB[16];
    __shared__ float dmat[GROUPS][TP * DSTR];   // WMMA D spill for diag extract

    // Decode block -> (n, s, h, w-tile)
    int b  = blockIdx.x;
    int wt = b & 15;  b >>= 4;          // W/16 = 16 tiles
    int h  = b & 63;  b >>= 6;          // H = 64
    int s  = b % S;
    int n  = b / S;

    const int lane  = threadIdx.x;      // wave32: 0..31
    const int p     = lane & 15;        // pixel within tile (= WMMA row & col)
    const int khalf = lane >> 4;        // which K-half this lane covers
    const int w_pix = wt * TP + p;

    // ---- zero the A-fragment pad runs + B zero block (one-time, 16B stores) ----
    {
        const v8h z8 = {};
        #pragma unroll
        for (int i = 0; i < 8; ++i) {
            const int r  = lane + 32 * i;          // 256 pad runs of 16B
            const int pp = r >> 4;
            const int gg = (r >> 1) & 7;
            const int hf = r & 1;
            *reinterpret_cast<v8h*>(&aL[pp][gg * 32 + 16 + hf * 8]) = z8;
        }
        if (lane < 2)
            *reinterpret_cast<v8h*>(&zB[lane * 8]) = z8;
    }

    // ---- per-pixel bilinear setup (done once, shared by 128-channel loop) ----
    const int eobase = ((n * (2 * S) + 2 * s) * HH + h) * WW + w_pix;
    const float eox = eo[eobase];
    const float eoy = eo[eobase + HH * WW];
    // base 3x3 offsets in 'ij' meshgrid order: base_x = s/3 - 1, base_y = s%3 - 1
    const float x = (float)w_pix + (float)(s / 3 - 1) + eox;
    const float y = (float)h     + (float)(s % 3 - 1) + eoy;

    const float x0f = floorf(x), y0f = floorf(y);
    const float fx = x - x0f, fy = y - y0f;
    const int ix0 = (int)x0f, iy0 = (int)y0f;
    const int ix1 = ix0 + 1,  iy1 = iy0 + 1;

    // zero-pad-outside semantics (pad-by-1 + clamp in the reference)
    const bool vx0 = (unsigned)ix0 < (unsigned)WW;
    const bool vx1 = (unsigned)ix1 < (unsigned)WW;
    const bool vy0 = (unsigned)iy0 < (unsigned)HH;
    const bool vy1 = (unsigned)iy1 < (unsigned)HH;

    float w00 = (1.f - fx) * (1.f - fy);  // (x0,y0)
    float w01 = (1.f - fx) * fy;          // (x0,y1)
    float w10 = fx * (1.f - fy);          // (x1,y0)
    float w11 = fx * fy;                  // (x1,y1)
    w00 = (vx0 && vy0) ? w00 : 0.f;
    w01 = (vx0 && vy1) ? w01 : 0.f;
    w10 = (vx1 && vy0) ? w10 : 0.f;
    w11 = (vx1 && vy1) ? w11 : 0.f;

    // clamped (always in-bounds) corner offsets; invalid corners weigh 0
    const int cx0 = min(max(ix0, 0), WW - 1);
    const int cx1 = min(max(ix1, 0), WW - 1);
    const int cy0 = min(max(iy0, 0), HH - 1);
    const int cy1 = min(max(iy1, 0), HH - 1);
    const int o00 = cy0 * WW + cx0;
    const int o01 = cy1 * WW + cx0;
    const int o10 = cy0 * WW + cx1;
    const int o11 = cy1 * WW + cx1;

    const float* rbase = right + (size_t)n * CC * HH * WW;
    const float* lbase = left  + (size_t)n * CC * HH * WW + h * WW + w_pix;

    // ---- gather phase: half-wave interleaves channels for its pixel ----
    #pragma unroll 4
    for (int c = khalf; c < CC; c += 2) {
        const float* rc = rbase + c * (HH * WW);
        const float v = w00 * rc[o00] + w01 * rc[o01]
                      + w10 * rc[o10] + w11 * rc[o11];
        bL[p][c] = (_Float16)v;
        // remap channel into A-fragment layout: cl -> cl (contiguous)
        const int cl = c & 15;
        aL[p][(c >> 4) * 32 + cl] = (_Float16)lbase[c * (HH * WW)];
    }
    __syncthreads();

    // ---- correlation phase: 8 WMMAs, diag(D) = per-pixel group dot ----
    const _Float16* zb = zB;
    #pragma unroll
    for (int g = 0; g < GROUPS; ++g) {
        // A fragment: one aligned 32B load, pads already zero in LDS
        const v16h a = *reinterpret_cast<const v16h*>(&aL[p][g * 32 + khalf * 16]);
        // B fragment: lanes 0-15 real columns, lanes 16-31 zero rows (broadcast)
        const _Float16* bp = (lane < 16) ? &bL[p][g * CG] : zb;
        const v16h bm = *reinterpret_cast<const v16h*>(bp);

        v8f acc = {};
        acc = __builtin_amdgcn_wmma_f32_16x16x32_f16(
            /*neg_a=*/false, a, /*neg_b=*/false, bm,
            /*c_mod=*/(short)0, acc, /*reuse_a=*/false, /*reuse_b=*/false);

        // D layout: lanes 0-15: VGPR r -> (M=r, N=lane); lanes 16-31: (M=8+r, N=lane-16)
        const int mb = khalf * 8;
        #pragma unroll
        for (int r = 0; r < 8; ++r)
            dmat[g][(mb + r) * DSTR + p] = acc[r];
    }
    __syncthreads();

    // ---- diagonal extract + store: corr[n, g*9+s, h, w] = diag/16 ----
    if (lane < 16) {
        float* obase = out + ((size_t)n * (GROUPS * S) + s) * HH * WW + h * WW + w_pix;
        #pragma unroll
        for (int g = 0; g < GROUPS; ++g)
            obase[g * (S * HH * WW)] = dmat[g][p * DSTR + p] * (1.0f / 16.0f);
    }
}

extern "C" void kernel_launch(void* const* d_in, const int* in_sizes, int n_in,
                              void* d_out, int out_size, void* d_ws, size_t ws_size,
                              hipStream_t stream)
{
    const float* left  = (const float*)d_in[0];
    const float* right = (const float*)d_in[1];
    const float* eo    = (const float*)d_in[2];
    float* out = (float*)d_out;

    const int blocks = NN * S * HH * (WW / TP);   // 2*9*64*16 = 18432 waves
    deform_corr_wmma<<<blocks, 32, 0, stream>>>(left, right, eo, out);
}
